// MultiHeadCausalSelfAttention_76476187672711
// MI455X (gfx1250) — matrix operand order repaired and checked by reading, not compiled, not run on hardware
//
#include <hip/hip_runtime.h>
#include <hip/hip_bf16.h>

// ---------------------------------------------------------------------------
// MI455X (gfx1250) multi-head causal self-attention, bf16 WMMA path with
// async-to-LDS staging (ASYNCcnt) and double-buffered GEMM tiles.
// B=4, T=2048, D=1024, H=16, Hd=64.
// ---------------------------------------------------------------------------

#define B_   4
#define T_   2048
#define D_   1024
#define H_   16
#define HD_  64
#define M_   (B_ * T_)      // 8192 rows for all projection GEMMs

typedef __attribute__((ext_vector_type(16))) __bf16 v16bf;
typedef __attribute__((ext_vector_type(8)))  float  v8f;

union Frag {                 // one WMMA bf16 A/B operand (8 VGPRs)
    v16bf v;
    unsigned int u[8];
    uint4 q[2];
};

static __device__ __forceinline__ unsigned short f2bf(float f) {
    unsigned int u = __float_as_uint(f);
    u += 0x7FFFu + ((u >> 16) & 1u);           // round-to-nearest-even
    return (unsigned short)(u >> 16);
}

static __device__ __forceinline__ v8f wmma_bf16(const v16bf& a, const v16bf& b, v8f c) {
    return __builtin_amdgcn_wmma_f32_16x16x32_bf16(
        /*neg_a*/false, a, /*neg_b*/false, b,
        /*c_mod*/(short)0, c, /*reuse_a*/false, /*reuse_b*/false);
}

// one async 16-byte copy global -> LDS (per lane), ASYNCcnt-tracked
static __device__ __forceinline__ void async_b128(unsigned ldsAddr, unsigned long long gAddr) {
    asm volatile("global_load_async_to_lds_b128 %0, %1, off"
                 :: "v"(ldsAddr), "v"(gAddr) : "memory");
}
static __device__ __forceinline__ void async_b128_x2(unsigned ldsAddr, unsigned long long gAddr) {
    asm volatile("global_load_async_to_lds_b128 %0, %1, off\n\t"
                 "global_load_async_to_lds_b128 %0, %1, off offset:16"
                 :: "v"(ldsAddr), "v"(gAddr) : "memory");
}
static __device__ __forceinline__ void async_b128_x4(unsigned ldsAddr, unsigned long long gAddr) {
    asm volatile("global_load_async_to_lds_b128 %0, %1, off\n\t"
                 "global_load_async_to_lds_b128 %0, %1, off offset:16\n\t"
                 "global_load_async_to_lds_b128 %0, %1, off offset:32\n\t"
                 "global_load_async_to_lds_b128 %0, %1, off offset:48"
                 :: "v"(ldsAddr), "v"(gAddr) : "memory");
}
static __device__ __forceinline__ unsigned lds_addr(const void* p) {
    return (unsigned)(size_t)p;   // low 32 bits of flat LDS-aperture address
}

// ---------------------------------------------------------------------------
// Kernel 1: f32 -> bf16 elementwise convert
// ---------------------------------------------------------------------------
__global__ __launch_bounds__(256) void cvt_f32_bf16(const float* __restrict__ in,
                                                    unsigned short* __restrict__ out,
                                                    int n) {
    int i = blockIdx.x * 256 + threadIdx.x;
    if (i < n) out[i] = f2bf(in[i]);
}

// ---------------------------------------------------------------------------
// Kernel 2: weight convert + transpose  W[k][n] (f32) -> Wt[n][k] (bf16)
// ---------------------------------------------------------------------------
__global__ __launch_bounds__(256) void cvt_w_transpose(const float* __restrict__ W,
                                                       unsigned short* __restrict__ Wt) {
    int i = blockIdx.x * 256 + threadIdx.x;   // 1M elements
    int k = i >> 10, n = i & 1023;
    Wt[n * 1024 + k] = f2bf(W[i]);
}

// ---------------------------------------------------------------------------
// Kernel 3: GEMM  C[M,N] = A[M,K] * Wt[N,K]^T, bf16 in / f32 accum.
// Double-buffered LDS tiles staged with global_load_async_to_lds_b128.
//   mode 0: write bf16 to QKV layout [B,H,T,64], scaled by outScale
//   mode 1: write f32  to out[row*1024+col] + bias[col]
// Block tile 128x128, 8 waves (4x2) of 32x64, K-step 32.
// ---------------------------------------------------------------------------
__global__ __launch_bounds__(256) void gemm_bf16(const unsigned short* __restrict__ A,
                                                 const unsigned short* __restrict__ Wt,
                                                 unsigned short* __restrict__ outBf,
                                                 float* __restrict__ outF,
                                                 const float* __restrict__ bias,
                                                 float outScale, int mode) {
    __shared__ __align__(16) unsigned short Atile[2][128 * 32];
    __shared__ __align__(16) unsigned short Btile[2][128 * 32];

    const int tid   = threadIdx.x;
    const int wave  = tid >> 5;
    const int lane  = tid & 31;
    const int half  = lane >> 4;
    const int ln    = lane & 15;
    const int waveM = wave >> 1;        // 0..3
    const int waveN = wave & 1;         // 0..1
    const int mBase = blockIdx.x * 128;
    const int nBase = blockIdx.y * 128;

    v8f acc[2][4] = {};

    const int srow = tid >> 1;          // staging row 0..127
    const int shk  = tid & 1;           // staging k-half
    const size_t aRowOff = (size_t)(mBase + srow) * 1024 + shk * 16;
    const size_t bRowOff = (size_t)(nBase + srow) * 1024 + shk * 16;

    auto issue = [&](int buf, int kk) {
        unsigned la = lds_addr(&Atile[buf][srow * 32 + shk * 16]);
        unsigned lb = lds_addr(&Btile[buf][srow * 32 + shk * 16]);
        async_b128_x2(la, (unsigned long long)(A  + aRowOff + kk));
        async_b128_x2(lb, (unsigned long long)(Wt + bRowOff + kk));
    };

    issue(0, 0);                         // prologue: chunk 0 -> buffer 0

    for (int c = 0; c < 32; ++c) {
        const int cur = c & 1;
        if (c + 1 < 32) {
            issue(cur ^ 1, (c + 1) * 32);                      // prefetch next chunk
            asm volatile("s_wait_asynccnt 4" ::: "memory");    // oldest 4 (current) done
        } else {
            asm volatile("s_wait_asynccnt 0" ::: "memory");
        }
        __syncthreads();

        const uint4* A4 = (const uint4*)Atile[cur];   // 4 uint4 per 32-elem row
        const uint4* B4 = (const uint4*)Btile[cur];

        Frag af[2];
        #pragma unroll
        for (int mt = 0; mt < 2; ++mt) {
            int row = waveM * 32 + mt * 16 + ln;
            af[mt].q[0] = A4[row * 4 + half];
            af[mt].q[1] = A4[row * 4 + half + 2];
        }
        Frag bf[4];
        #pragma unroll
        for (int nt = 0; nt < 4; ++nt) {
            int nRow = waveN * 64 + nt * 16 + ln;
            bf[nt].q[0] = B4[nRow * 4 + half];
            bf[nt].q[1] = B4[nRow * 4 + half + 2];
        }
        #pragma unroll
        for (int mt = 0; mt < 2; ++mt)
            #pragma unroll
            for (int nt = 0; nt < 4; ++nt)
                acc[mt][nt] = wmma_bf16(af[mt].v, bf[nt].v, acc[mt][nt]);

        __syncthreads();   // all waves done reading cur before it is re-staged
    }

    // epilogue
    #pragma unroll
    for (int mt = 0; mt < 2; ++mt) {
        #pragma unroll
        for (int nt = 0; nt < 4; ++nt) {
            int col = nBase + waveN * 64 + nt * 16 + ln;
            #pragma unroll
            for (int r = 0; r < 8; ++r) {
                int row = mBase + waveM * 32 + mt * 16 + r + 8 * half;
                float v = acc[mt][nt][r];
                if (mode == 0) {
                    int b = row >> 11, t = row & 2047;
                    int h = col >> 6,  d = col & 63;
                    outBf[(((size_t)(b * H_ + h) * T_ + t) << 6) + d] = f2bf(v * outScale);
                } else {
                    outF[(size_t)row * 1024 + col] = v + bias[col];
                }
            }
        }
    }
}

// ---------------------------------------------------------------------------
// Kernel 4: flash attention per (b, h, 128-row q tile). Each wave owns a
// 16-row q strip; kv processed in chunks of 32 staged through LDS (K/Q via
// async-to-LDS). Q is pre-scaled by 1/sqrt(64) at projection time.
// ---------------------------------------------------------------------------
__global__ __launch_bounds__(256) void attn_flash(const unsigned short* __restrict__ Q,
                                                  const unsigned short* __restrict__ K,
                                                  const unsigned short* __restrict__ V,
                                                  unsigned short* __restrict__ ctx) {
    __shared__ __align__(16) unsigned short Qlds[128 * 64];
    __shared__ __align__(16) unsigned short Klds[32 * 64];   // [kv][d]  (B for QK^T)
    __shared__ __align__(16) unsigned short Vtld[64 * 32];   // [d][kv]  (B for PV)
    __shared__ __align__(16) unsigned short Plds[8 * 16 * 32];

    const int tid  = threadIdx.x;
    const int wave = tid >> 5;
    const int lane = tid & 31;
    const int half = lane >> 4;
    const int ln   = lane & 15;

    const int qTile = blockIdx.x;      // 0..15
    const int h     = blockIdx.y;      // 0..15
    const int b     = blockIdx.z;      // 0..3
    const int qBase = qTile * 128;

    const size_t headOff = (size_t)(b * H_ + h) * T_ * HD_;
    const unsigned short* Qh = Q + headOff;
    const unsigned short* Kh = K + headOff;
    const unsigned short* Vh = V + headOff;

    // stage full Q tile 128x64 via async copies (64 B per thread)
    {
        int row = tid >> 1, hh = tid & 1;
        async_b128_x4(lds_addr(&Qlds[row * 64 + hh * 32]),
                      (unsigned long long)(Qh + (size_t)(qBase + row) * 64 + hh * 32));
    }
    asm volatile("s_wait_asynccnt 0" ::: "memory");
    __syncthreads();

    // per-wave persistent Q fragments (16 rows x 64 d, two K-chunks of 32)
    Frag qf[2];
    {
        const uint4* Q4 = (const uint4*)Qlds;    // 8 uint4 per 64-elem row
        int row = wave * 16 + ln;
        #pragma unroll
        for (int kc = 0; kc < 2; ++kc) {
            qf[kc].q[0] = Q4[row * 8 + kc * 4 + half];
            qf[kc].q[1] = Q4[row * 8 + kc * 4 + half + 2];
        }
    }

    v8f o[4] = {};
    float mrow[8], lrow[8];
    #pragma unroll
    for (int r = 0; r < 8; ++r) { mrow[r] = -1e30f; lrow[r] = 0.0f; }

    const int myQstart = qBase + wave * 16;
    const int myQend   = myQstart + 15;
    const int nChunks  = (qBase + 128) >> 5;

    for (int c = 0; c < nChunks; ++c) {
        const int kvBase = c * 32;
        __syncthreads();
        { // stage K chunk [32][64] async; V chunk transposed [64][32] manually
            int row = tid >> 3, seg = tid & 7;
            async_b128(lds_addr(&Klds[row * 64 + seg * 8]),
                       (unsigned long long)(Kh + (size_t)(kvBase + row) * 64 + seg * 8));
            uint4 vv = *((const uint4*)(Vh + (size_t)(kvBase + row) * 64 + seg * 8));
            const unsigned short* ve = (const unsigned short*)&vv;
            #pragma unroll
            for (int j = 0; j < 8; ++j)
                Vtld[(seg * 8 + j) * 32 + row] = ve[j];
        }
        asm volatile("s_wait_asynccnt 0" ::: "memory");
        __syncthreads();

        if (kvBase <= myQend) {
            const uint4* K4 = (const uint4*)Klds;   // 8 uint4 per 64-elem row
            const uint4* V4 = (const uint4*)Vtld;   // 4 uint4 per 32-elem row

            // S = Q K^T for 16 rows x 32 kv cols (two 16x16 tiles)
            v8f s[2] = {};
            #pragma unroll
            for (int nt = 0; nt < 2; ++nt) {
                int kvLocal = nt * 16 + ln;
                #pragma unroll
                for (int kc = 0; kc < 2; ++kc) {
                    Frag bk;
                    bk.q[0] = K4[kvLocal * 8 + kc * 4 + half];
                    bk.q[1] = K4[kvLocal * 8 + kc * 4 + half + 2];
                    s[nt] = wmma_bf16(qf[kc].v, bk.v, s[nt]);
                }
            }

            // online softmax (row stats live per lane-half, rows r + 8*half)
            const int col0 = kvBase + ln;
            const int col1 = kvBase + 16 + ln;
            #pragma unroll
            for (int r = 0; r < 8; ++r) {
                int qrow = myQstart + r + 8 * half;
                bool ok0 = (col0 <= qrow), ok1 = (col1 <= qrow);
                float s0 = ok0 ? s[0][r] : -1e30f;
                float s1 = ok1 ? s[1][r] : -1e30f;
                float cm = fmaxf(s0, s1);
                #pragma unroll
                for (int off = 8; off >= 1; off >>= 1)
                    cm = fmaxf(cm, __shfl_xor(cm, off, 32));
                float mnew  = fmaxf(mrow[r], cm);
                float p0 = ok0 ? __expf(s0 - mnew) : 0.0f;
                float p1 = ok1 ? __expf(s1 - mnew) : 0.0f;
                float cs = p0 + p1;
                #pragma unroll
                for (int off = 8; off >= 1; off >>= 1)
                    cs += __shfl_xor(cs, off, 32);
                float scale = __expf(mrow[r] - mnew);
                lrow[r] = lrow[r] * scale + cs;
                mrow[r] = mnew;
                #pragma unroll
                for (int t = 0; t < 4; ++t) o[t][r] = o[t][r] * scale;
                // P -> LDS (C layout -> memory row-major) for A-frag rebuild
                int prow = r + 8 * half;
                Plds[wave * 512 + prow * 32 + ln]      = f2bf(p0);
                Plds[wave * 512 + prow * 32 + 16 + ln] = f2bf(p1);
            }
            asm volatile("s_wait_dscnt 0" ::: "memory");

            // P (16x32) as A fragment, then O += P @ V
            Frag pf;
            {
                const uint4* P4 = (const uint4*)Plds;  // 4 uint4 per 32-elem row
                pf.q[0] = P4[wave * 64 + ln * 4 + half];
                pf.q[1] = P4[wave * 64 + ln * 4 + half + 2];
            }
            #pragma unroll
            for (int nt = 0; nt < 4; ++nt) {
                int dRow = nt * 16 + ln;
                Frag bv;
                bv.q[0] = V4[dRow * 4 + half];
                bv.q[1] = V4[dRow * 4 + half + 2];
                o[nt] = wmma_bf16(pf.v, bv.v, o[nt]);
            }
        }
    }

    // normalize and write ctx [B,T,D] bf16
    #pragma unroll
    for (int r = 0; r < 8; ++r) {
        float inv = 1.0f / lrow[r];
        int qrow = myQstart + r + 8 * half;
        #pragma unroll
        for (int nt = 0; nt < 4; ++nt) {
            int d = nt * 16 + ln;
            ctx[((size_t)(b * T_ + qrow)) * D_ + h * HD_ + d] = f2bf(o[nt][r] * inv);
        }
    }
}

// ---------------------------------------------------------------------------
// Host-side orchestration
// ---------------------------------------------------------------------------
extern "C" void kernel_launch(void* const* d_in, const int* in_sizes, int n_in,
                              void* d_out, int out_size, void* d_ws, size_t ws_size,
                              hipStream_t stream) {
    const float* x  = (const float*)d_in[0];
    const float* Wq = (const float*)d_in[1];
    const float* Wk = (const float*)d_in[2];
    const float* Wv = (const float*)d_in[3];
    const float* Wo = (const float*)d_in[4];
    const float* bo = (const float*)d_in[5];
    float* out = (float*)d_out;

    char* ws = (char*)d_ws;
    const size_t SZ_X   = (size_t)M_ * D_ * 2;       // 16 MB bf16
    const size_t SZ_W   = (size_t)D_ * D_ * 2;       // 2 MB bf16
    unsigned short* xbf = (unsigned short*)(ws);
    unsigned short* wqt = (unsigned short*)(ws + SZ_X);
    unsigned short* wkt = (unsigned short*)(ws + SZ_X + 1 * SZ_W);
    unsigned short* wvt = (unsigned short*)(ws + SZ_X + 2 * SZ_W);
    unsigned short* wot = (unsigned short*)(ws + SZ_X + 3 * SZ_W);
    unsigned short* qws = (unsigned short*)(ws + SZ_X + 4 * SZ_W);
    unsigned short* kws = (unsigned short*)(ws + SZ_X + 4 * SZ_W + 1 * SZ_X);
    unsigned short* vws = (unsigned short*)(ws + SZ_X + 4 * SZ_W + 2 * SZ_X);
    unsigned short* cws = (unsigned short*)(ws + SZ_X + 4 * SZ_W + 3 * SZ_X);

    // 1) converts
    cvt_f32_bf16<<<(M_ * D_) / 256, 256, 0, stream>>>(x, xbf, M_ * D_);
    cvt_w_transpose<<<(D_ * D_) / 256, 256, 0, stream>>>(Wq, wqt);
    cvt_w_transpose<<<(D_ * D_) / 256, 256, 0, stream>>>(Wk, wkt);
    cvt_w_transpose<<<(D_ * D_) / 256, 256, 0, stream>>>(Wv, wvt);
    cvt_w_transpose<<<(D_ * D_) / 256, 256, 0, stream>>>(Wo, wot);

    // 2) projections (softmax scale folded into Q)
    dim3 gGemm(M_ / 128, D_ / 128);
    const float qscale = 0.125f;   // 1/sqrt(64)
    gemm_bf16<<<gGemm, 256, 0, stream>>>(xbf, wqt, qws, nullptr, nullptr, qscale, 0);
    gemm_bf16<<<gGemm, 256, 0, stream>>>(xbf, wkt, kws, nullptr, nullptr, 1.0f, 0);
    gemm_bf16<<<gGemm, 256, 0, stream>>>(xbf, wvt, vws, nullptr, nullptr, 1.0f, 0);

    // 3) flash attention
    dim3 gAttn(T_ / 128, H_, B_);
    attn_flash<<<gAttn, 256, 0, stream>>>(qws, kws, vws, cws);

    // 4) output projection + bias -> f32
    gemm_bf16<<<gGemm, 256, 0, stream>>>(cws, wot, nullptr, out, bo, 1.0f, 1);
}
